// actor_13743895347442
// MI455X (gfx1250) — matrix operand order repaired and checked by reading, not compiled, not run on hardware
//
#include <hip/hip_runtime.h>
#include <hip/hip_bf16.h>
#include <stdint.h>

// ---------------- problem constants ----------------
constexpr int N_NODES  = 51200;
constexpr int N_GRAPHS = 256;
constexpr int CITIES   = N_NODES / N_GRAPHS;   // 200
constexpr int N_EDGES  = 819200;
constexpr int H        = 128;
constexpr int H2       = 256;
constexpr int H4       = 512;

// ---------------- workspace layout (bytes) ----------------
constexpr size_t OFF_STATS = 0;                          // 256 f32 (sum[128], sumsq[128])
constexpr size_t OFF_WT    = 4096;                       // transposed bf16 weights
constexpr size_t OFF_geW1t = OFF_WT;                     // 128*128*2 = 32768
constexpr size_t OFF_geW2t = OFF_geW1t + 32768;
constexpr size_t OFF_gW1t  = OFF_geW2t + 32768;          // 256*256*2 = 131072
constexpr size_t OFF_gW2t  = OFF_gW1t  + 131072;
constexpr size_t OFF_l1Wt  = OFF_gW2t  + 131072;         // 512*128*2 = 131072
constexpr size_t OFF_R1 = 524288;                        // f32 scratch: x0raw / agg / logits
constexpr size_t SZ_R1  = (size_t)N_NODES * H2 * 4;      // 52428800
constexpr size_t OFF_R2 = OFF_R1 + SZ_R1;                // bf16 N x 256
constexpr size_t SZ_R2  = (size_t)N_NODES * H2 * 2;
constexpr size_t OFF_R3 = OFF_R2 + SZ_R2;                // bf16 N x 256
constexpr size_t OFF_R4 = OFF_R3 + SZ_R2;                // bf16 N x 512
// total ws used = OFF_R4 + N*512*2 = 157,810,688 bytes

// ---------------- types / helpers ----------------
typedef __attribute__((ext_vector_type(16))) __bf16 v16bf;
typedef __attribute__((ext_vector_type(8)))  float  v8f;

__device__ __forceinline__ unsigned short f2b(float x) {
  union { float f; uint32_t u; } c; c.f = x;
  uint32_t u = c.u;
  uint32_t r = u + 0x7FFFu + ((u >> 16) & 1u);   // round-to-nearest-even
  return (unsigned short)(r >> 16);
}
__device__ __forceinline__ float b2f(unsigned short b) {
  union { uint32_t u; float f; } c; c.u = ((uint32_t)b) << 16;
  return c.f;
}

// ---------------- utility kernels ----------------
__global__ void zero_f32(float* p, size_t n) {
  size_t i = (size_t)blockIdx.x * blockDim.x + threadIdx.x;
  if (i < n) p[i] = 0.0f;
}

// transpose f32 W[K x M] -> bf16 Wt[M x K]
__global__ void tcast_kernel(const float* __restrict__ W, unsigned short* __restrict__ Wt,
                             int K, int M) {
  int i = blockIdx.x * blockDim.x + threadIdx.x;
  if (i >= K * M) return;
  int k = i / M, m = i - k * M;
  Wt[(size_t)m * K + k] = f2b(W[i]);
}

// ---------------- stage 1: linear0 + BN stats ----------------
__global__ void lin0_stats_kernel(const float* __restrict__ x, const float* __restrict__ W0,
                                  const float* __restrict__ b0, float* __restrict__ x0raw,
                                  float* __restrict__ stats) {
  __shared__ float sh[256], sh2[256];
  int f   = threadIdx.x & 127;
  int sub = threadIdx.x >> 7;            // 0 or 1
  int rbase = blockIdx.x * 128;
  float w0 = W0[f], w1 = W0[128 + f], bb = b0[f];
  float s = 0.f, ss = 0.f;
  for (int r = rbase + sub; r < rbase + 128; r += 2) {
    float v = x[2 * r] * w0 + x[2 * r + 1] * w1 + bb;
    x0raw[(size_t)r * H + f] = v;
    s += v; ss += v * v;
  }
  sh[threadIdx.x] = s; sh2[threadIdx.x] = ss;
  __syncthreads();
  if (sub == 0) {
    atomicAdd(&stats[f],        s  + sh[128 + f]);
    atomicAdd(&stats[128 + f],  ss + sh2[128 + f]);
  }
}

__global__ void bn_apply_kernel(const float* __restrict__ x0raw, const float* __restrict__ stats,
                                const float* __restrict__ gamma, const float* __restrict__ beta,
                                unsigned short* __restrict__ h0) {
  size_t i = (size_t)blockIdx.x * blockDim.x + threadIdx.x;
  if (i >= (size_t)N_NODES * H) return;
  int f = (int)(i & (H - 1));
  float mu  = stats[f] * (1.0f / N_NODES);
  float var = stats[H + f] * (1.0f / N_NODES) - mu * mu;
  float v = (x0raw[i] - mu) * rsqrtf(var + 1e-5f) * gamma[f] + beta[f];
  h0[i] = f2b(v);
}

// ---------------- edge segment-sum: agg[dst] += h[src] ----------------
// one wave per edge; each lane: one 64-bit load (4 contiguous bf16) + 4 f32 atomics.
__global__ void edge_agg_kernel(const unsigned short* __restrict__ hb,
                                const int* __restrict__ src, const int* __restrict__ dst,
                                float* __restrict__ agg, int D) {
  int e = blockIdx.x * (blockDim.x >> 5) + (threadIdx.x >> 5);
  if (e >= N_EDGES) return;
  int lane = threadIdx.x & 31;
  const unsigned short* hr = hb + (size_t)src[e] * D;
  float* ar = agg + (size_t)dst[e] * D;
  for (int base = lane * 4; base < D; base += 128) {
    uint2 p = *(const uint2*)(hr + base);   // 4 bf16
    atomicAdd(&ar[base + 0], b2f((unsigned short)(p.x & 0xffffu)));
    atomicAdd(&ar[base + 1], b2f((unsigned short)(p.x >> 16)));
    atomicAdd(&ar[base + 2], b2f((unsigned short)(p.y & 0xffffu)));
    atomicAdd(&ar[base + 3], b2f((unsigned short)(p.y >> 16)));
  }
}

// z = h + agg (bf16 out)
__global__ void add_cast_kernel(const unsigned short* __restrict__ hb,
                                const float* __restrict__ agg,
                                unsigned short* __restrict__ zb, size_t n) {
  size_t i = (size_t)blockIdx.x * blockDim.x + threadIdx.x;
  if (i < n) zb[i] = f2b(b2f(hb[i]) + agg[i]);
}

// ---------------- WMMA GEMM: out = act(A[N x K] @ W[K x M] + bias) ----------------
// A row-major bf16; Wt is W transposed (M x K) row-major bf16.
// block = 128 threads (4 waves); each wave: one 16-row tile x four 16-col tiles.
// B tile (64 cols x <=256 K-chunk) staged in LDS, fixed padded stride so every
// DS address folds into the immediate offset field. K, M compile-time.
template <int K, int M>
__global__ void wmma_gemm_kernel(const unsigned short* __restrict__ A,
                                 const unsigned short* __restrict__ Wt,
                                 const float* __restrict__ bias,
                                 unsigned short* __restrict__ outb,
                                 float* __restrict__ outf,
                                 int relu) {
  constexpr int STRIDE = 264;                       // 528B rows: 16B aligned, bank-skewed
  constexpr int KC     = (K < 256) ? K : 256;       // chunk (K always multiple of KC)
  __shared__ unsigned short Bs[64 * STRIDE];

  int wave = threadIdx.x >> 5;
  int lane = threadIdx.x & 31;
  int lr = lane & 15, lh = lane >> 4;
  int row0 = (blockIdx.x * 4 + wave) * 16;
  int col0 = blockIdx.y * 64;

  v8f acc[4];
  #pragma unroll
  for (int t = 0; t < 4; ++t) acc[t] = (v8f){0.f,0.f,0.f,0.f,0.f,0.f,0.f,0.f};

  const unsigned short* Arow = A + (size_t)(row0 + lr) * K;
  const unsigned short* Bbase = Bs + lr * STRIDE + lh * 16;   // + const per (t,kk,v)

  #pragma unroll
  for (int kc = 0; kc < K; kc += KC) {
    // cooperative stage of B tile: rows = output cols col0..col0+63
    {
      int r    = threadIdx.x >> 1;               // 0..63
      int half = threadIdx.x & 1;
      constexpr int per = KC >> 1;               // 64 or 128 elements
      const unsigned short* sp = Wt + (size_t)(col0 + r) * K + kc + half * per;
      unsigned short* dp = Bs + r * STRIDE + half * per;
      #pragma unroll
      for (int c = 0; c < per; c += 8)
        *(uint4*)(dp + c) = *(const uint4*)(sp + c);
    }
    __syncthreads();

    #pragma unroll
    for (int kk = 0; kk < KC; kk += 32) {
      // A fragment (16x32 bf16): lanes 0-15 K=0..15 ; lanes 16-31 K=16..31
      union { uint32_t u[8]; v16bf v; } af;
      #pragma unroll
      for (int v = 0; v < 8; ++v) {
        int kb = lh * 16 + v * 2;
        af.u[v] = *(const uint32_t*)(Arow + kc + kk + kb);
      }
      // all four B fragments first (independent regs -> WMMAs issue back-to-back)
      union { uint32_t u[8]; v16bf v; } bf[4];
      #pragma unroll
      for (int t = 0; t < 4; ++t)
        #pragma unroll
        for (int v = 0; v < 8; ++v)
          bf[t].u[v] = *(const uint32_t*)(Bbase + t * 16 * STRIDE + kk + 2 * v);
      #pragma unroll
      for (int t = 0; t < 4; ++t)
        acc[t] = __builtin_amdgcn_wmma_f32_16x16x32_bf16(
            false, af.v, false, bf[t].v, (short)0, acc[t], false, false);
    }
    __syncthreads();
  }

  #pragma unroll
  for (int t = 0; t < 4; ++t) {
    int n = col0 + t * 16 + lr;
    float bv = bias[n];
    #pragma unroll
    for (int v = 0; v < 8; ++v) {
      int m = row0 + v + lh * 8;
      float val = acc[t][v] + bv;
      if (relu) val = fmaxf(val, 0.0f);
      size_t o = (size_t)m * M + n;
      if (outf) outf[o] = val;
      outb[o] = f2b(val);
    }
  }
}

// ---------------- concat builders ----------------
__global__ void build_xin_kernel(const float* __restrict__ init_embed,
                                 const unsigned short* __restrict__ x0b,
                                 unsigned short* __restrict__ xin) {
  size_t i = (size_t)blockIdx.x * blockDim.x + threadIdx.x;
  if (i >= (size_t)N_NODES * H2) return;
  int c = (int)(i & (H2 - 1));
  size_t r = i >> 8;
  xin[i] = (c < H) ? f2b(init_embed[c]) : x0b[r * H + (c - H)];
}

__global__ void build_cat3_kernel(const unsigned short* __restrict__ enc,
                                  const int* __restrict__ center,
                                  unsigned short* __restrict__ cat) {
  size_t i = (size_t)blockIdx.x * blockDim.x + threadIdx.x;
  if (i >= (size_t)N_NODES * H4) return;
  int c = (int)(i & (H4 - 1));
  size_t r = i >> 9;
  int g = (int)(r / CITIES);
  cat[i] = (c < H2) ? enc[(size_t)center[g] * H2 + c] : enc[r * H2 + (c - H2)];
}

// ---------------- final 130-wide dot -> logits ----------------
__global__ void logits_kernel(const float* __restrict__ x, const unsigned short* __restrict__ hf,
                              const float* __restrict__ l2W, const float* __restrict__ l2b,
                              const unsigned char* __restrict__ mask,
                              float* __restrict__ logits) {
  int r = blockIdx.x * blockDim.x + threadIdx.x;
  if (r >= N_NODES) return;
  float acc = x[2 * r] * l2W[0] + x[2 * r + 1] * l2W[1];
  const unsigned short* h = hf + (size_t)r * H;
  #pragma unroll 4
  for (int f = 0; f < H; ++f) acc += b2f(h[f]) * l2W[2 + f];
  acc += l2b[0];
  if (!mask[r]) acc -= 1.0e6f;
  logits[r] = acc;
}

// ---------------- per-graph softmax + argmax + log-prob ----------------
__global__ void softmax_select_kernel(const float* __restrict__ logits,
                                      const int* __restrict__ gid,
                                      float* __restrict__ out_sample,
                                      float* __restrict__ out_log) {
  __shared__ float sv[256];
  __shared__ int   si[256];
  int g = blockIdx.x, t = threadIdx.x;
  float v = (t < CITIES) ? logits[g * CITIES + t] : -3.4e38f;
  sv[t] = v; __syncthreads();
  for (int s = 128; s > 0; s >>= 1) { if (t < s) sv[t] = fmaxf(sv[t], sv[t + s]); __syncthreads(); }
  float m = sv[0]; __syncthreads();
  float e = (t < CITIES) ? __expf(v - m) : 0.0f;
  sv[t] = e; __syncthreads();
  for (int s = 128; s > 0; s >>= 1) { if (t < s) sv[t] += sv[t + s]; __syncthreads(); }
  float S = sv[0]; __syncthreads();
  sv[t] = v; si[t] = t; __syncthreads();
  for (int s = 128; s > 0; s >>= 1) {
    if (t < s) {
      float vo = sv[t + s]; int io = si[t + s];
      if (vo > sv[t] || (vo == sv[t] && io < si[t])) { sv[t] = vo; si[t] = io; }
    }
    __syncthreads();
  }
  if (t == 0) {
    out_sample[g] = (float)(gid[g] + si[0]);
    out_log[g]    = (sv[0] - m) - logf(S);
  }
}

// ---------------- launch ----------------
extern "C" void kernel_launch(void* const* d_in, const int* in_sizes, int n_in,
                              void* d_out, int out_size, void* d_ws, size_t ws_size,
                              hipStream_t stream) {
  (void)in_sizes; (void)n_in; (void)out_size; (void)ws_size;
  const float* x        = (const float*)d_in[0];
  const int*   eidx     = (const int*)d_in[1];
  const int*   src      = eidx;
  const int*   dst      = eidx + N_EDGES;
  const int*   center   = (const int*)d_in[3];
  const unsigned char* mask = (const unsigned char*)d_in[4];
  const int*   gid      = (const int*)d_in[5];
  const float* W0    = (const float*)d_in[6];
  const float* b0    = (const float*)d_in[7];
  const float* gamma = (const float*)d_in[8];
  const float* beta  = (const float*)d_in[9];
  const float* geW1  = (const float*)d_in[10];
  const float* geb1  = (const float*)d_in[11];
  const float* geW2  = (const float*)d_in[12];
  const float* geb2  = (const float*)d_in[13];
  const float* iemb  = (const float*)d_in[14];
  const float* gW1   = (const float*)d_in[15];
  const float* gb1   = (const float*)d_in[16];
  const float* gW2   = (const float*)d_in[17];
  const float* gb2   = (const float*)d_in[18];
  const float* l1W   = (const float*)d_in[19];
  const float* l1b   = (const float*)d_in[20];
  const float* l2W   = (const float*)d_in[21];
  const float* l2b   = (const float*)d_in[22];

  char* ws = (char*)d_ws;
  float*          stats = (float*)(ws + OFF_STATS);
  unsigned short* geW1t = (unsigned short*)(ws + OFF_geW1t);
  unsigned short* geW2t = (unsigned short*)(ws + OFF_geW2t);
  unsigned short* gW1t  = (unsigned short*)(ws + OFF_gW1t);
  unsigned short* gW2t  = (unsigned short*)(ws + OFF_gW2t);
  unsigned short* l1Wt  = (unsigned short*)(ws + OFF_l1Wt);
  float*          R1f   = (float*)(ws + OFF_R1);
  unsigned short* R2    = (unsigned short*)(ws + OFF_R2);
  unsigned short* R3    = (unsigned short*)(ws + OFF_R3);
  unsigned short* R4    = (unsigned short*)(ws + OFF_R4);

  float* out_sample = (float*)d_out;
  float* out_log    = out_sample + N_GRAPHS;
  float* out_x0     = out_log + N_GRAPHS;   // [N,128] f32

  const int B = 256;
  dim3 blk(B);

  // weight transposes (bf16)
  tcast_kernel<<<(H * H + B - 1) / B, blk, 0, stream>>>(geW1, geW1t, H, H);
  tcast_kernel<<<(H * H + B - 1) / B, blk, 0, stream>>>(geW2, geW2t, H, H);
  tcast_kernel<<<(H2 * H2 + B - 1) / B, blk, 0, stream>>>(gW1, gW1t, H2, H2);
  tcast_kernel<<<(H2 * H2 + B - 1) / B, blk, 0, stream>>>(gW2, gW2t, H2, H2);
  tcast_kernel<<<(H4 * H + B - 1) / B, blk, 0, stream>>>(l1W, l1Wt, H4, H);

  // stage 1: linear0 + batchnorm
  zero_f32<<<1, blk, 0, stream>>>(stats, 256);
  lin0_stats_kernel<<<N_NODES / 128, blk, 0, stream>>>(x, W0, b0, R1f, stats);
  size_t nH = (size_t)N_NODES * H;
  bn_apply_kernel<<<(unsigned)((nH + B - 1) / B), blk, 0, stream>>>(R1f, stats, gamma, beta, R2);

  // GIN 1 (128-d): agg, z1, two GEMMs
  zero_f32<<<(unsigned)((nH + B - 1) / B), blk, 0, stream>>>(R1f, nH);
  edge_agg_kernel<<<N_EDGES / 8, blk, 0, stream>>>(R2, src, dst, R1f, H);
  add_cast_kernel<<<(unsigned)((nH + B - 1) / B), blk, 0, stream>>>(R2, R1f, R3, nH);
  wmma_gemm_kernel<H, H><<<dim3(N_NODES / 64, H / 64), dim3(128), 0, stream>>>(
      R3, geW1t, geb1, R2, nullptr, 1);
  wmma_gemm_kernel<H, H><<<dim3(N_NODES / 64, H / 64), dim3(128), 0, stream>>>(
      R2, geW2t, geb2, R3, out_x0, 0);   // x0: f32 to d_out + bf16

  // x_in = [init_embed | x0]  (N x 256)
  size_t nH2 = (size_t)N_NODES * H2;
  build_xin_kernel<<<(unsigned)((nH2 + B - 1) / B), blk, 0, stream>>>(iemb, R3, R4);

  // GIN 2 (256-d)
  zero_f32<<<(unsigned)((nH2 + B - 1) / B), blk, 0, stream>>>(R1f, nH2);
  edge_agg_kernel<<<N_EDGES / 8, blk, 0, stream>>>(R4, src, dst, R1f, H2);
  add_cast_kernel<<<(unsigned)((nH2 + B - 1) / B), blk, 0, stream>>>(R4, R1f, R2, nH2);
  wmma_gemm_kernel<H2, H2><<<dim3(N_NODES / 64, H2 / 64), dim3(128), 0, stream>>>(
      R2, gW1t, gb1, R3, nullptr, 1);
  wmma_gemm_kernel<H2, H2><<<dim3(N_NODES / 64, H2 / 64), dim3(128), 0, stream>>>(
      R3, gW2t, gb2, R2, nullptr, 0);    // enc -> R2

  // cat3 = [enc[center] | enc]  (N x 512), then h = relu(cat3 @ l1W + l1b)
  size_t nH4 = (size_t)N_NODES * H4;
  build_cat3_kernel<<<(unsigned)((nH4 + B - 1) / B), blk, 0, stream>>>(R2, center, R4);
  wmma_gemm_kernel<H4, H><<<dim3(N_NODES / 64, H / 64), dim3(128), 0, stream>>>(
      R4, l1Wt, l1b, R3, nullptr, 1);    // h -> R3 (N x 128)

  // logits + per-graph softmax/argmax
  logits_kernel<<<N_NODES / B, blk, 0, stream>>>(x, R3, l2W, l2b, mask, R1f);
  softmax_select_kernel<<<N_GRAPHS, blk, 0, stream>>>(R1f, gid, out_sample, out_log);
}
